// GAT_GCN_72679436582986
// MI455X (gfx1250) — matrix operand order repaired and checked by reading, not compiled, not run on hardware
//
#include <hip/hip_runtime.h>
#include <hip/hip_bf16.h>

typedef __attribute__((ext_vector_type(16))) _Float16 v16h;
typedef __attribute__((ext_vector_type(8)))  _Float16 v8h;
typedef __attribute__((ext_vector_type(8)))  float    v8f;

#define N_NODES 32768
#define N_EDGES 131072
#define ET      (N_EDGES + N_NODES)   // self loops appended
#define BG      512
#define F_IN    78
#define HEADS   10
#define HID     780

// float atomic max via signed/unsigned integer atomics (valid for all finite floats)
__device__ __forceinline__ void atomicMaxF(float* addr, float v) {
    if (v >= 0.f) atomicMax((int*)addr, __float_as_int(v));
    else          atomicMin((unsigned int*)addr, __float_as_uint(v));
}

// =====================================================================
// Generic WMMA-f16 GEMM: C[M,N] = act(A[M,K] @ B[K,N] + bias)
// block = 256 threads (8 waves), block tile 128x64, K-step 32.
// Software-pipelined: next K-tile's global loads are issued between the
// two barriers and consumed one iteration later, hiding memory latency
// behind the WMMA work.
// Requirements (all call sites satisfy): M % 128 == 0, K even, N even,
// lda/ldb even (pairwise 8B-aligned staging).
// =====================================================================
#define BM 128
#define BN 64
#define BK 32
#define AS_LD 40   // halfs; 80 B row stride (16B aligned, bank-spread)
#define BS_LD 40

__global__ __launch_bounds__(256) void gemm_wmma_kernel(
    const float* __restrict__ A, const float* __restrict__ B,
    const float* __restrict__ bias, float* __restrict__ C,
    int M, int N, int K, int lda, int ldb, int ldc, int relu)
{
    __shared__ __align__(16) _Float16 As [BM * AS_LD];   // [m][k]
    __shared__ __align__(16) _Float16 BsT[BN * BS_LD];   // [n][k] (transposed)
    const int tid  = threadIdx.x;
    const int lane = tid & 31;
    const int wave = tid >> 5;
    const int bm   = blockIdx.y * BM;
    const int bn   = blockIdx.x * BN;
    const int m0   = wave * 16;
    const int nl   = lane & 15;
    const int hi   = lane >> 4;          // 0 or 1 (lane half)

    // per-thread fixed staging coordinates
    const int am = (tid >> 4);           // A rows this thread touches: am, am+16, ...
    const int ak = (tid & 15) * 2;       // A k-pair (even)
    const int bk = (tid >> 5);           // B k rows: bk, bk+8, bk+16, bk+24
    const int bnp = (tid & 31) * 2;      // B n-pair (even)

    v8f acc0 = {}, acc1 = {}, acc2 = {}, acc3 = {};
    float2 ra[8];                        // A tile pairs in flight
    float2 rb[4];                        // B tile pairs in flight

    // ---- prologue: issue loads for first K-tile ----
#pragma unroll
    for (int it = 0; it < 8; ++it) {
        int gk  = ak;                                   // kt = 0
        int gks = gk < K ? gk : K - 2;
        ra[it] = *(const float2*)&A[(size_t)(bm + am + it * 16) * lda + gks];
    }
#pragma unroll
    for (int it = 0; it < 4; ++it) {
        int gk  = bk + it * 8;
        int gks = gk < K ? gk : K - 1;
        int gn  = bn + bnp;
        int gns = gn < N ? gn : N - 2;
        rb[it] = *(const float2*)&B[(size_t)gks * ldb + gns];
    }

    for (int kt = 0; kt < K; kt += BK) {
        // ---- store phase: convert in-flight registers into LDS ----
#pragma unroll
        for (int it = 0; it < 8; ++it) {
            bool ok = (kt + ak) < K;
            int m = am + it * 16;
            As[m * AS_LD + ak]     = ok ? (_Float16)ra[it].x : (_Float16)0.f;
            As[m * AS_LD + ak + 1] = ok ? (_Float16)ra[it].y : (_Float16)0.f;
        }
#pragma unroll
        for (int it = 0; it < 4; ++it) {
            int k = bk + it * 8;
            bool ok = ((kt + k) < K) && ((bn + bnp) < N);
            BsT[bnp * BS_LD + k]       = ok ? (_Float16)rb[it].x : (_Float16)0.f;
            BsT[(bnp + 1) * BS_LD + k] = ok ? (_Float16)rb[it].y : (_Float16)0.f;
        }
        __syncthreads();

        // ---- issue next K-tile's global loads (consumed next iteration) ----
        int ktn = kt + BK;
        if (ktn < K) {
#pragma unroll
            for (int it = 0; it < 8; ++it) {
                int gk  = ktn + ak;
                int gks = gk < K ? gk : K - 2;
                ra[it] = *(const float2*)&A[(size_t)(bm + am + it * 16) * lda + gks];
            }
#pragma unroll
            for (int it = 0; it < 4; ++it) {
                int gk  = ktn + bk + it * 8;
                int gks = gk < K ? gk : K - 1;
                int gn  = bn + bnp;
                int gns = gn < N ? gn : N - 2;
                rb[it] = *(const float2*)&B[(size_t)gks * ldb + gns];
            }
        }

        // ---- A fragment: two contiguous 8-half groups -> 2x ds_load_b128 ----
        const _Float16* ap = As + (m0 + nl) * AS_LD + (hi << 3);
        v8h alo = *(const v8h*)ap;          // k = base+0..7
        v8h ahi = *(const v8h*)(ap + 16);   // k = base+16..23
        v16h af = __builtin_shufflevector(alo, ahi, 0,1,2,3,4,5,6,7,8,9,10,11,12,13,14,15);

        // ---- B fragments: two 8-half groups per lane (k = kh+0..7, kh+16..23) -> 2x ds_load_b128 each ----
        const int kh = hi << 3;             // 0 or 8
#pragma unroll
        for (int t = 0; t < 4; ++t) {
            const _Float16* bp = BsT + (t * 16 + nl) * BS_LD + kh;
            v8h blo = *(const v8h*)bp;
            v8h bhi = *(const v8h*)(bp + 16);
            v16h bf = __builtin_shufflevector(blo, bhi, 0,1,2,3,4,5,6,7,8,9,10,11,12,13,14,15);
            if      (t == 0) acc0 = __builtin_amdgcn_wmma_f32_16x16x32_f16(false, af, false, bf, (short)0, acc0, false, false);
            else if (t == 1) acc1 = __builtin_amdgcn_wmma_f32_16x16x32_f16(false, af, false, bf, (short)0, acc1, false, false);
            else if (t == 2) acc2 = __builtin_amdgcn_wmma_f32_16x16x32_f16(false, af, false, bf, (short)0, acc2, false, false);
            else             acc3 = __builtin_amdgcn_wmma_f32_16x16x32_f16(false, af, false, bf, (short)0, acc3, false, false);
        }
        __syncthreads();
    }

    // ---- writeback: C/D layout: vgpr r -> M = r + 8*hi; N = lane&15 ----
#pragma unroll
    for (int t = 0; t < 4; ++t) {
        const v8f* accp = (t == 0) ? &acc0 : (t == 1) ? &acc1 : (t == 2) ? &acc2 : &acc3;
        int n = bn + t * 16 + nl;
        if (n < N) {
            float badd = bias ? bias[n] : 0.f;
#pragma unroll
            for (int r = 0; r < 8; ++r) {
                int m = bm + m0 + r + (hi << 3);
                if (m < M) {
                    float v = (*accp)[r] + badd;
                    if (relu) v = fmaxf(v, 0.f);
                    C[(size_t)m * ldc + n] = v;
                }
            }
        }
    }
}

// =====================================================================
// Small helper kernels
// =====================================================================
__global__ void fill_kernel(float* p, float v, int n) {
    int i = blockIdx.x * 256 + threadIdx.x;
    if (i < n) p[i] = v;
}

// per-(node,head) attention projections: a = sum_c h[n,h,c]*att[h,c]
__global__ void att_proj_kernel(const float* __restrict__ h,
                                const float* __restrict__ att_src,
                                const float* __restrict__ att_dst,
                                float* __restrict__ a_src, float* __restrict__ a_dst)
{
    int idx = blockIdx.x * 256 + threadIdx.x;
    if (idx >= N_NODES * HEADS) return;
    int n = idx / HEADS, hh = idx % HEADS;
    const float* hp = h + (size_t)n * HID + hh * F_IN;
    const float* ws = att_src + hh * F_IN;
    const float* wd = att_dst + hh * F_IN;
    float s1 = 0.f, s2 = 0.f;
    for (int c = 0; c < F_IN; ++c) { float v = hp[c]; s1 += v * ws[c]; s2 += v * wd[c]; }
    a_src[idx] = s1; a_dst[idx] = s2;
}

__device__ __forceinline__ void edge_sd(const int* ei, int e, int& s, int& d) {
    if (e < N_EDGES) { s = ei[e]; d = ei[N_EDGES + e]; }
    else             { s = d = e - N_EDGES; }
}

__global__ void edge_max_kernel(const int* __restrict__ ei,
                                const float* __restrict__ a_src,
                                const float* __restrict__ a_dst,
                                float* __restrict__ amax)
{
    int e = blockIdx.x * 256 + threadIdx.x;
    if (e >= ET) return;
    int s, d; edge_sd(ei, e, s, d);
    for (int hh = 0; hh < HEADS; ++hh) {
        float al = a_src[s * HEADS + hh] + a_dst[d * HEADS + hh];
        al = al > 0.f ? al : 0.2f * al;            // leaky_relu 0.2
        atomicMaxF(&amax[d * HEADS + hh], al);
    }
}

__global__ void edge_denom_kernel(const int* __restrict__ ei,
                                  const float* __restrict__ a_src,
                                  const float* __restrict__ a_dst,
                                  const float* __restrict__ amax,
                                  float* __restrict__ denom, float* __restrict__ deg)
{
    int e = blockIdx.x * 256 + threadIdx.x;
    if (e >= ET) return;
    int s, d; edge_sd(ei, e, s, d);
    atomicAdd(&deg[d], 1.f);                        // GCN in-degree (incl self loop)
    for (int hh = 0; hh < HEADS; ++hh) {
        float al = a_src[s * HEADS + hh] + a_dst[d * HEADS + hh];
        al = al > 0.f ? al : 0.2f * al;
        atomicAdd(&denom[d * HEADS + hh], expf(al - amax[d * HEADS + hh]));
    }
}

// g1[dst,h,c] += h[src,h,c] * softmax_weight(e,h)   (one block per edge)
__global__ __launch_bounds__(256) void gat_agg_kernel(
    const int* __restrict__ ei, const float* __restrict__ h,
    const float* __restrict__ a_src, const float* __restrict__ a_dst,
    const float* __restrict__ amax, const float* __restrict__ denom,
    float* __restrict__ g1)
{
    int e = blockIdx.x;
    int s, d; edge_sd(ei, e, s, d);
    __shared__ float wgt[HEADS];
    if (threadIdx.x < HEADS) {
        int hh = threadIdx.x;
        float al = a_src[s * HEADS + hh] + a_dst[d * HEADS + hh];
        al = al > 0.f ? al : 0.2f * al;
        wgt[hh] = expf(al - amax[d * HEADS + hh]) / denom[d * HEADS + hh];
    }
    __syncthreads();
    const float* hs = h + (size_t)s * HID;
    float* gd = g1 + (size_t)d * HID;
    for (int i = threadIdx.x; i < HID; i += 256)
        atomicAdd(&gd[i], hs[i] * wgt[i / F_IN]);
}

__global__ void bias_relu_kernel(float* __restrict__ buf, const float* __restrict__ bias,
                                 int total, int cols)
{
    int i = blockIdx.x * 256 + threadIdx.x;
    if (i >= total) return;
    buf[i] = fmaxf(buf[i] + bias[i % cols], 0.f);
}

// g2[dst,:] += z[src,:] * rsqrt(deg[src])*rsqrt(deg[dst])
__global__ __launch_bounds__(256) void gcn_agg_kernel(
    const int* __restrict__ ei, const float* __restrict__ z,
    const float* __restrict__ deg, float* __restrict__ g2)
{
    int e = blockIdx.x;
    int s, d; edge_sd(ei, e, s, d);
    float nrm = rsqrtf(deg[s]) * rsqrtf(deg[d]);
    const float* zs = z + (size_t)s * HID;
    float* gd = g2 + (size_t)d * HID;
    for (int i = threadIdx.x; i < HID; i += 256)
        atomicAdd(&gd[i], zs[i] * nrm);
}

// global max + mean pool: batch = node/64 (64 contiguous nodes per graph)
__global__ void pool_kernel(const float* __restrict__ g2, float* __restrict__ xg)
{
    int b = blockIdx.x;
    for (int col = threadIdx.x; col < HID; col += 256) {
        const float* base = g2 + (size_t)(b * 64) * HID + col;
        float mx = -1e30f, sm = 0.f;
        for (int j = 0; j < 64; ++j) { float v = base[(size_t)j * HID]; mx = fmaxf(mx, v); sm += v; }
        xg[b * (2 * HID) + col]       = mx;
        xg[b * (2 * HID) + HID + col] = sm * (1.f / 64.f);
    }
}

// Protein branch: conv1d(1000->32, k=8) over embedded targets, restructured:
//   A[t,o,k] = sum_{ci: target[b,ci]==t} w[o,ci,k]   (26x32x8 in LDS)
//   out[b,o,p] = bias[o] + sum_{t,k} emb[t,p+k] * A[t,o,k]
__global__ __launch_bounds__(256) void protein_conv_kernel(
    const int* __restrict__ target, const float* __restrict__ w,
    const float* __restrict__ cb, const float* __restrict__ emb,
    float* __restrict__ out)
{
    __shared__ float Al[26 * 256];     // [t][o*8+k]
    __shared__ float embs[26 * 128];
    int b = blockIdx.x, tid = threadIdx.x;
    for (int i = tid; i < 26 * 256; i += 256) Al[i] = 0.f;
    for (int i = tid; i < 26 * 128; i += 256) embs[i] = emb[i];
    __syncthreads();
    int o = tid >> 3, k = tid & 7;                 // thread owns (o,k)
    const float* wok = w + (size_t)o * 8000 + k;   // w[o][ci][k], OIH
    const int* tb = target + b * 1000;
    for (int ci = 0; ci < 1000; ++ci) {
        int t = tb[ci];
        Al[t * 256 + tid] += wok[ci * 8];
    }
    __syncthreads();
    for (int idx = tid; idx < 32 * 121; idx += 256) {
        int oo = idx / 121, p = idx - oo * 121;
        float acc = cb[oo];
        for (int t = 0; t < 26; ++t) {
            const float* ep = embs + t * 128 + p;
            const float* aa = Al + t * 256 + oo * 8;
#pragma unroll
            for (int kk = 0; kk < 8; ++kk) acc += ep[kk] * aa[kk];
        }
        out[(size_t)b * 3872 + idx] = acc;         // matches reshape(B,32*121)
    }
}

__global__ void head_out_kernel(const float* __restrict__ t3, const float* __restrict__ ow,
                                const float* __restrict__ ob, float* __restrict__ out)
{
    int b = blockIdx.x * 256 + threadIdx.x;
    if (b >= BG) return;
    const float* r = t3 + (size_t)b * 512;
    float acc = ob[0];
    for (int kx = 0; kx < 512; ++kx) acc += r[kx] * ow[kx];
    out[b] = acc;
}

// =====================================================================
extern "C" void kernel_launch(void* const* d_in, const int* in_sizes, int n_in,
                              void* d_out, int out_size, void* d_ws, size_t ws_size,
                              hipStream_t stream)
{
    (void)in_sizes; (void)n_in; (void)out_size; (void)ws_size;
    const float* x       = (const float*)d_in[0];
    const int*   ei      = (const int*)d_in[1];
    /* d_in[2] = batch (arange//64, implicit in pool kernel) */
    const int*   target  = (const int*)d_in[3];
    const float* gat_w   = (const float*)d_in[4];
    const float* att_src = (const float*)d_in[5];
    const float* att_dst = (const float*)d_in[6];
    const float* gat_b   = (const float*)d_in[7];
    const float* gcn_w   = (const float*)d_in[8];
    const float* gcn_b   = (const float*)d_in[9];
    const float* fcg1_w  = (const float*)d_in[10];
    const float* fcg1_b  = (const float*)d_in[11];
    const float* fcg2_w  = (const float*)d_in[12];
    const float* fcg2_b  = (const float*)d_in[13];
    const float* emb     = (const float*)d_in[14];
    const float* cvw     = (const float*)d_in[15];
    const float* cvb     = (const float*)d_in[16];
    const float* fcxt_w  = (const float*)d_in[17];
    const float* fcxt_b  = (const float*)d_in[18];
    const float* fc1_w   = (const float*)d_in[19];
    const float* fc1_b   = (const float*)d_in[20];
    const float* fc2_w   = (const float*)d_in[21];
    const float* fc2_b   = (const float*)d_in[22];
    const float* out_w   = (const float*)d_in[23];
    const float* out_b   = (const float*)d_in[24];
    float* out = (float*)d_out;

    // ---- workspace carve-up (floats); big node buffers are reused ----
    float* ws = (float*)d_ws;
    const size_t NB = (size_t)N_NODES * HID;        // 25,559,040 floats
    float* bufA   = ws;                             // h, then z
    float* bufB   = bufA + NB;                      // g1, then g2
    float* a_src  = bufB + NB;
    float* a_dst  = a_src + (size_t)N_NODES * HEADS;
    float* amax   = a_dst + (size_t)N_NODES * HEADS;
    float* denom  = amax + (size_t)N_NODES * HEADS;
    float* deg    = denom + (size_t)N_NODES * HEADS;
    float* xg     = deg + N_NODES;                  // [512,1560]
    float* t1     = xg + (size_t)BG * 2 * HID;      // [512,1500]
    float* xc     = t1 + (size_t)BG * 1500;         // [512,256]
    float* xt_pre = xc + (size_t)BG * 256;          // [512,3872]
    float* t2     = xt_pre + (size_t)BG * 3872;     // [512,1024]
    float* t3     = t2 + (size_t)BG * 1024;         // [512,512]

    dim3 blk(256);

    // ---- init accumulators ----
    hipMemsetAsync(denom, 0, (size_t)N_NODES * HEADS * sizeof(float), stream);
    hipMemsetAsync(deg,   0, (size_t)N_NODES * sizeof(float), stream);
    hipMemsetAsync(bufB,  0, NB * sizeof(float), stream);   // g1 accumulator
    fill_kernel<<<(N_NODES * HEADS + 255) / 256, blk, 0, stream>>>(amax, -1e30f, N_NODES * HEADS);

    // ---- GAT: h = x @ gat_w ----
    { dim3 g((HID + BN - 1) / BN, N_NODES / BM);
      gemm_wmma_kernel<<<g, blk, 0, stream>>>(x, gat_w, nullptr, bufA,
                                              N_NODES, HID, F_IN, F_IN, HID, HID, 0); }
    att_proj_kernel<<<(N_NODES * HEADS + 255) / 256, blk, 0, stream>>>(bufA, att_src, att_dst, a_src, a_dst);
    edge_max_kernel<<<(ET + 255) / 256, blk, 0, stream>>>(ei, a_src, a_dst, amax);
    edge_denom_kernel<<<(ET + 255) / 256, blk, 0, stream>>>(ei, a_src, a_dst, amax, denom, deg);
    gat_agg_kernel<<<ET, blk, 0, stream>>>(ei, bufA, a_src, a_dst, amax, denom, bufB);
    bias_relu_kernel<<<((int)NB + 255) / 256, blk, 0, stream>>>(bufB, gat_b, (int)NB, HID);

    // ---- GCN: z = g1 @ gcn_w (z overwrites h), then normalized scatter ----
    { dim3 g((HID + BN - 1) / BN, N_NODES / BM);
      gemm_wmma_kernel<<<g, blk, 0, stream>>>(bufB, gcn_w, nullptr, bufA,
                                              N_NODES, HID, HID, HID, HID, HID, 0); }
    hipMemsetAsync(bufB, 0, NB * sizeof(float), stream);    // g2 accumulator
    gcn_agg_kernel<<<ET, blk, 0, stream>>>(ei, bufA, deg, bufB);
    bias_relu_kernel<<<((int)NB + 255) / 256, blk, 0, stream>>>(bufB, gcn_b, (int)NB, HID);

    // ---- pooling -> xg [512,1560] ----
    pool_kernel<<<BG, blk, 0, stream>>>(bufB, xg);

    // ---- graph head: fcg1 (relu) -> fcg2 -> xc[:, :128] ----
    { dim3 g((1500 + BN - 1) / BN, BG / BM);
      gemm_wmma_kernel<<<g, blk, 0, stream>>>(xg, fcg1_w, fcg1_b, t1,
                                              BG, 1500, 2 * HID, 2 * HID, 1500, 1500, 1); }
    { dim3 g((128 + BN - 1) / BN, BG / BM);
      gemm_wmma_kernel<<<g, blk, 0, stream>>>(t1, fcg2_w, fcg2_b, xc,
                                              BG, 128, 1500, 1500, 128, 256, 0); }

    // ---- protein branch: symbol-bucketed conv, then fcxt -> xc[:, 128:] ----
    protein_conv_kernel<<<BG, blk, 0, stream>>>(target, cvw, cvb, emb, xt_pre);
    { dim3 g((128 + BN - 1) / BN, BG / BM);
      gemm_wmma_kernel<<<g, blk, 0, stream>>>(xt_pre, fcxt_w, fcxt_b, xc + 128,
                                              BG, 128, 3872, 3872, 128, 256, 0); }

    // ---- final MLP head ----
    { dim3 g((1024 + BN - 1) / BN, BG / BM);
      gemm_wmma_kernel<<<g, blk, 0, stream>>>(xc, fc1_w, fc1_b, t2,
                                              BG, 1024, 256, 256, 1024, 1024, 1); }
    { dim3 g((512 + BN - 1) / BN, BG / BM);
      gemm_wmma_kernel<<<g, blk, 0, stream>>>(t2, fc2_w, fc2_b, t3,
                                              BG, 512, 1024, 1024, 512, 512, 1); }
    head_out_kernel<<<(BG + 255) / 256, blk, 0, stream>>>(t3, out_w, out_b, out);
}
